// GCGRUCell_time_19473381720836
// MI455X (gfx1250) — matrix operand order repaired and checked by reading, not compiled, not run on hardware
//
#include <hip/hip_runtime.h>
#include <cstddef>

// ---------------- problem constants ----------------
#define BSZ    4096
#define DIN    32
#define DMODEL 64
#define NNODE  36
#define FFDIM  64
#define SP     48            // padded sequence length per batch element
#define BBB    4             // batch elements per workgroup
#define RTOK   (BBB * SP)    // 192 padded token rows per workgroup
#define CINZ   384           // (ORDER+1) * (DIN + DMODEL)
#define NTHR   512           // 16 wave32s -> 4 waves per SIMD (1 WG/WGP by LDS)
#define NWAVE  16

typedef __attribute__((ext_vector_type(16))) __bf16 v16bf;
typedef __attribute__((ext_vector_type(8)))  float  v8f;
typedef unsigned short u16;
typedef unsigned int   u32;
typedef __attribute__((ext_vector_type(8)))  u32 v8u;
typedef __attribute__((ext_vector_type(16))) u16 v16u;

// ---------------- bf16 conversions (native hardware cvt) ----------------
__device__ __forceinline__ u16 f2b(float f) {
  return __builtin_bit_cast(u16, (__bf16)f);
}
__device__ __forceinline__ float b2f(u16 u) {
  return (float)__builtin_bit_cast(__bf16, u);
}
__device__ __forceinline__ u32 f2b2(float lo, float hi) {  // packed pair
  return (u32)f2b(lo) | ((u32)f2b(hi) << 16);
}

// ---------------- WMMA wrapper ----------------
__device__ __forceinline__ v8f wmma32(v16bf a, v16bf b, v8f c) {
  // D(16x16,f32) = A(16x32,bf16) x B(32x16,bf16) + C
  return __builtin_amdgcn_wmma_f32_16x16x32_bf16(false, a, false, b,
                                                 (short)0, c, false, false);
}

// A fragment: 16x32 tile from row-major [rows][stride]; K pairs contiguous
// -> 8 aligned dword LDS loads, no packing, no branches.
__device__ __forceinline__ v16bf ldA(const u16* base, int stride, int m0, int k0,
                                     int lane) {
  const u16* row = base + (size_t)(m0 + (lane & 15)) * stride;
  const int kb = k0 + ((lane & 16) ? 16 : 0);
  v8u d;
#pragma unroll
  for (int j = 0; j < 8; ++j) {
    int k = kb + 2 * j;       // even, dword aligned
    d[j] = *reinterpret_cast<const u32*>(row + k);
  }
  return __builtin_bit_cast(v16bf, d);
}

// B fragment where memory holds B^T row-major: mem[n][k] (K-contiguous)
__device__ __forceinline__ v16bf ldBT(const u16* base, int stride, int k0, int n0,
                                      int lane) {
  const u32* row = reinterpret_cast<const u32*>(
      base + (size_t)(n0 + (lane & 15)) * stride + k0 + ((lane & 16) ? 16 : 0));
  v8u d;
#pragma unroll
  for (int j = 0; j < 8; ++j) d[j] = row[j];
  return __builtin_bit_cast(v16bf, d);
}

// B fragment from row-major [K][stride] (strided; used only for gate-GEMM z)
__device__ __forceinline__ v16bf ldBs(const u16* base, int stride, int k0, int n0,
                                      int lane) {
  const u16* col = base + (n0 + (lane & 15));
  const int kb = k0 + ((lane & 16) ? 16 : 0);
  v16u d;
#pragma unroll
  for (int i = 0; i < 16; ++i) d[i] = col[(size_t)(kb + i) * stride];
  return __builtin_bit_cast(v16bf, d);
}

// ---------------- LDS layout (union of sequential stages) ----------------
struct SharedA {                       // transformer stage
  u16   qk[RTOK][128];                 // 49152 B : q(0..63), k(64..127)
  u16   vT[BBB][64][64];               // 32768 B : v transposed [d][t], t>=48 zero
  float scores[BBB][SP][SP];           // 36864 B
  u16   attnP[BBB][SP][SP];            // 18432 B
  u16   attnP_pad[32];                 //    64 B zero pad (read-past safety)
  u16   tmp[RTOK][DMODEL];             // 24576 B
  float scratch[RTOK][DMODEL];         // 49152 B
  u16   w_in[3 * DMODEL][DMODEL];      // 24576 B : natural [o][d] layout
  u16   w_out[DMODEL][DMODEL];         //  8192 B
  u16   w_ff1[FFDIM][DMODEL];          //  8192 B
  u16   w_ff2[DMODEL][FFDIM];          //  8192 B
};                                     // 260160 B
struct SharedD {                       // diffusion-conv + gate stage
  u16   z[BBB][CINZ][SP];              // 147456 B (channel-major per bb)
  u16   adjP[SP][64];                  //   6144 B : adj[w][v], zero outside 36x36
  float h2[BBB][DMODEL][NNODE];        //  36864 B
  u16   uw[DMODEL][CINZ];              //  49152 B (natural [o][c] layout)
  u16   cw[DMODEL][CINZ];              //  49152 B
};                                     // 288768 B
struct Shared {
  u16 tok[RTOK][DMODEL];               // 24576 B, persists across stages
  union { SharedA a; SharedD d; } u;
};                                     // total 313344 B (<= 320 KB/WG)

struct Params {
  const float *x, *dt, *h, *adj;
  const float *in_w[2], *in_b[2], *out_w[2], *out_b[2];
  const float *ff1_w[2], *ff1_b[2], *ff2_w[2], *ff2_b[2];
  const float *ln1_g[2], *ln1_b[2], *ln2_g[2], *ln2_b[2];
  const float *u_w, *u_b, *c_w, *c_b;
  float *out;
};

__global__ __launch_bounds__(NTHR, 1)
void gcgru_fused_kernel(Params p) {
  __shared__ Shared sh;
  const int tid  = threadIdx.x;
  const int lane = tid & 31;
  const int wave = tid >> 5;
  const int b0   = blockIdx.x * BBB;
  const int moff0 = (lane & 16) ? 8 : 0;   // D-matrix row offset for lane half

  // -------- Stage 0: tokens e[s][d] = h[b][d][s], pad rows zeroed --------
  for (int i = tid; i < RTOK * DMODEL; i += NTHR) {
    int r = i >> 6, d = i & 63;
    int bb = r / SP, s = r % SP;
    float v = 0.f;
    if (s < NNODE) v = p.h[((size_t)(b0 + bb) * DMODEL + d) * NNODE + s];
    sh.tok[r][d] = f2b(v);
  }
  // zero attnP pad rows (s = 36..47) once; never written again
  for (int i = tid; i < BBB * 12 * SP; i += NTHR) {
    int bb = i / (12 * SP); int rem = i % (12 * SP);
    sh.u.a.attnP[bb][NNODE + rem / SP][rem % SP] = 0;
  }
  // zero vT padded columns t = 48..63 (never written by QKV epilogue)
  for (int i = tid; i < BBB * 64 * 16; i += NTHR) {
    int bb = i / (64 * 16); int rem = i % (64 * 16);
    sh.u.a.vT[bb][rem >> 4][48 + (rem & 15)] = 0;
  }
  if (tid < 32) sh.u.a.attnP_pad[tid] = 0;
  __syncthreads();

  // =================== two transformer encoder layers ===================
  for (int L = 0; L < 2; ++L) {
    // ---- preload weights (natural K-contiguous layouts) ----
    for (int i = tid; i < 3 * DMODEL * DMODEL; i += NTHR)
      sh.u.a.w_in[0][i] = f2b(p.in_w[L][i]);
    for (int i = tid; i < DMODEL * DMODEL; i += NTHR) {
      sh.u.a.w_out[0][i] = f2b(p.out_w[L][i]);
      sh.u.a.w_ff1[0][i] = f2b(p.ff1_w[L][i]);
      sh.u.a.w_ff2[0][i] = f2b(p.ff2_w[L][i]);
    }
    __syncthreads();

    // ---- Q,K projection tiles (nt 0..7): row-major store to qk ----
    for (int t = wave; t < 12 * 8; t += NWAVE) {
      int mt = t / 8, nt = t % 8;
      v8f acc = {};
#pragma unroll
      for (int kk = 0; kk < 64; kk += 32) {
        v16bf a = ldA(&sh.tok[0][0], DMODEL, mt * 16, kk, lane);
        v16bf b = ldBT(&sh.u.a.w_in[0][0], DMODEL, kk, nt * 16, lane);
        acc = wmma32(a, b, acc);
      }
      int n = nt * 16 + (lane & 15);
      float bias = p.in_b[L][n];
#pragma unroll
      for (int r = 0; r < 8; ++r)
        sh.u.a.qk[mt * 16 + r + moff0][n] = f2b(acc[r] + bias);
    }
    // ---- V projection tiles (nt 8..11): transposed, packed b128 store ----
    for (int t = wave; t < 12 * 4; t += NWAVE) {
      int mt = t / 4, nt = 8 + (t % 4);
      v8f acc = {};
#pragma unroll
      for (int kk = 0; kk < 64; kk += 32) {
        v16bf a = ldA(&sh.tok[0][0], DMODEL, mt * 16, kk, lane);
        v16bf b = ldBT(&sh.u.a.w_in[0][0], DMODEL, kk, nt * 16, lane);
        acc = wmma32(a, b, acc);
      }
      int n = nt * 16 + (lane & 15);
      float bias = p.in_b[L][n];
      int d = n - 128;
      int mbase = mt * 16 + moff0;       // 8-row block never crosses 48-boundary
      int bb = mbase / SP, s0 = mbase % SP;   // s0 multiple of 8 -> 16B aligned
      uint4 w;
      w.x = f2b2(acc[0] + bias, acc[1] + bias);
      w.y = f2b2(acc[2] + bias, acc[3] + bias);
      w.z = f2b2(acc[4] + bias, acc[5] + bias);
      w.w = f2b2(acc[6] + bias, acc[7] + bias);
      *reinterpret_cast<uint4*>(&sh.u.a.vT[bb][d][s0]) = w;
    }
    __syncthreads();

    // ---- scores = q k^T / 8 : per bb [48x64] x [64x48] ----
    for (int t = wave; t < BBB * 9; t += NWAVE) {
      int bb = t / 9, mt = (t % 9) / 3, nt = t % 3;
      v8f acc = {};
#pragma unroll
      for (int kk = 0; kk < 64; kk += 32) {
        v16bf a = ldA(&sh.u.a.qk[bb * SP][0], 128, mt * 16, kk, lane);
        // B[k=d][n=t'] = k-token(t')[d] -> memory qk[bb*SP+t'][64+d] is [n][k]
        v16bf b = ldBT(&sh.u.a.qk[bb * SP][64], 128, kk, nt * 16, lane);
        acc = wmma32(a, b, acc);
      }
      int n = nt * 16 + (lane & 15);
#pragma unroll
      for (int r = 0; r < 8; ++r)
        sh.u.a.scores[bb][mt * 16 + r + moff0][n] = acc[r] * 0.125f;
    }
    __syncthreads();

    // ---- softmax over t' < 36; write bf16 probs, zero padded cols ----
    for (int idx = tid; idx < BBB * NNODE; idx += NTHR) {
      int bb = idx / NNODE, s = idx % NNODE;
      const float* row = sh.u.a.scores[bb][s];
      float mx = row[0];
      for (int t2 = 1; t2 < NNODE; ++t2) mx = fmaxf(mx, row[t2]);
      float sum = 0.f;
      for (int t2 = 0; t2 < NNODE; ++t2) sum += __expf(row[t2] - mx);
      float inv = 1.f / sum;
      for (int t2 = 0; t2 < NNODE; ++t2)
        sh.u.a.attnP[bb][s][t2] = f2b(__expf(row[t2] - mx) * inv);
      for (int t2 = NNODE; t2 < SP; ++t2) sh.u.a.attnP[bb][s][t2] = 0;
    }
    __syncthreads();

    // ---- o = attn x V : per bb [48x48(Kpad64)] x [48x64] ----
    // A reads past row end for k>=48; product is 0 because vT[:,48..63]==0.
    for (int t = wave; t < BBB * 12; t += NWAVE) {
      int bb = t / 12, mt = (t % 12) / 4, nt = t % 4;
      v8f acc = {};
#pragma unroll
      for (int kk = 0; kk < 64; kk += 32) {
        v16bf a = ldA(&sh.u.a.attnP[bb][0][0], SP, mt * 16, kk, lane);
        v16bf b = ldBT(&sh.u.a.vT[bb][0][0], 64, kk, nt * 16, lane);
        acc = wmma32(a, b, acc);
      }
      int n = nt * 16 + (lane & 15);
#pragma unroll
      for (int r = 0; r < 8; ++r)
        sh.u.a.tmp[bb * SP + mt * 16 + r + moff0][n] = f2b(acc[r]);
    }
    __syncthreads();

    // ---- out-proj + residual -> scratch(f32) ----
    for (int t = wave; t < 12 * 4; t += NWAVE) {
      int mt = t / 4, nt = t % 4;
      v8f acc = {};
#pragma unroll
      for (int kk = 0; kk < 64; kk += 32) {
        v16bf a = ldA(&sh.u.a.tmp[0][0], DMODEL, mt * 16, kk, lane);
        v16bf b = ldBT(&sh.u.a.w_out[0][0], DMODEL, kk, nt * 16, lane);
        acc = wmma32(a, b, acc);
      }
      int n = nt * 16 + (lane & 15);
      float bias = p.out_b[L][n];
#pragma unroll
      for (int r = 0; r < 8; ++r) {
        int m = mt * 16 + r + moff0;
        sh.u.a.scratch[m][n] = acc[r] + bias + b2f(sh.tok[m][n]);
      }
    }
    __syncthreads();

    // ---- LN1 -> tok (pad rows stay exactly 0) ----
    for (int r = tid; r < RTOK; r += NTHR) {
      if ((r % SP) >= NNODE) {
        for (int d = 0; d < DMODEL; ++d) sh.tok[r][d] = 0;
        continue;
      }
      float mean = 0.f;
      for (int d = 0; d < DMODEL; ++d) mean += sh.u.a.scratch[r][d];
      mean *= (1.f / DMODEL);
      float var = 0.f;
      for (int d = 0; d < DMODEL; ++d) {
        float dd = sh.u.a.scratch[r][d] - mean; var += dd * dd;
      }
      float inv = rsqrtf(var * (1.f / DMODEL) + 1e-5f);
      for (int d = 0; d < DMODEL; ++d)
        sh.tok[r][d] = f2b((sh.u.a.scratch[r][d] - mean) * inv * p.ln1_g[L][d] +
                           p.ln1_b[L][d]);
    }
    __syncthreads();

    // ---- FF1 + ReLU -> tmp ----
    for (int t = wave; t < 12 * 4; t += NWAVE) {
      int mt = t / 4, nt = t % 4;
      v8f acc = {};
#pragma unroll
      for (int kk = 0; kk < 64; kk += 32) {
        v16bf a = ldA(&sh.tok[0][0], DMODEL, mt * 16, kk, lane);
        v16bf b = ldBT(&sh.u.a.w_ff1[0][0], DMODEL, kk, nt * 16, lane);
        acc = wmma32(a, b, acc);
      }
      int n = nt * 16 + (lane & 15);
      float bias = p.ff1_b[L][n];
#pragma unroll
      for (int r = 0; r < 8; ++r)
        sh.u.a.tmp[mt * 16 + r + moff0][n] = f2b(fmaxf(acc[r] + bias, 0.f));
    }
    __syncthreads();

    // ---- FF2 + residual -> scratch ----
    for (int t = wave; t < 12 * 4; t += NWAVE) {
      int mt = t / 4, nt = t % 4;
      v8f acc = {};
#pragma unroll
      for (int kk = 0; kk < 64; kk += 32) {
        v16bf a = ldA(&sh.u.a.tmp[0][0], FFDIM, mt * 16, kk, lane);
        v16bf b = ldBT(&sh.u.a.w_ff2[0][0], FFDIM, kk, nt * 16, lane);
        acc = wmma32(a, b, acc);
      }
      int n = nt * 16 + (lane & 15);
      float bias = p.ff2_b[L][n];
#pragma unroll
      for (int r = 0; r < 8; ++r) {
        int m = mt * 16 + r + moff0;
        sh.u.a.scratch[m][n] = acc[r] + bias + b2f(sh.tok[m][n]);
      }
    }
    __syncthreads();

    // ---- LN2 -> tok ----
    for (int r = tid; r < RTOK; r += NTHR) {
      if ((r % SP) >= NNODE) {
        for (int d = 0; d < DMODEL; ++d) sh.tok[r][d] = 0;
        continue;
      }
      float mean = 0.f;
      for (int d = 0; d < DMODEL; ++d) mean += sh.u.a.scratch[r][d];
      mean *= (1.f / DMODEL);
      float var = 0.f;
      for (int d = 0; d < DMODEL; ++d) {
        float dd = sh.u.a.scratch[r][d] - mean; var += dd * dd;
      }
      float inv = rsqrtf(var * (1.f / DMODEL) + 1e-5f);
      for (int d = 0; d < DMODEL; ++d)
        sh.tok[r][d] = f2b((sh.u.a.scratch[r][d] - mean) * inv * p.ln2_g[L][d] +
                           p.ln2_b[L][d]);
    }
    __syncthreads();
  }

  // ============== Stage C: h2, xg, weights into union D ==============
  for (int i = tid; i < SP * 64; i += NTHR) {        // adjP[w][v], zero pad
    int w = i >> 6, v = i & 63;
    sh.u.d.adjP[w][v] = f2b((w < NNODE && v < NNODE) ? p.adj[w * NNODE + v] : 0.f);
  }
  for (int i = tid; i < DMODEL * CINZ; i += NTHR) {  // gate weights, natural layout
    sh.u.d.uw[0][i] = f2b(p.u_w[i]);
    sh.u.d.cw[0][i] = f2b(p.c_w[i]);
  }
  for (int i = tid; i < BBB * 96 * SP; i += NTHR) {  // z[:, 0:96, :] and h2
    int bb = i / (96 * SP); int rem = i % (96 * SP);
    int c = rem / SP, n = rem % SP;
    float val = 0.f;
    if (n < NNODE) {
      if (c < DIN) {
        val = p.x[((size_t)(b0 + bb) * DIN + c) * NNODE + n];
      } else {
        int d = c - DIN;
        float dt = p.dt[b0 + bb];
        float eps = b2f(sh.tok[bb * SP + n][d]);
        float hv  = p.h[((size_t)(b0 + bb) * DMODEL + d) * NNODE + n];
        val = sqrtf(1.f - dt) * hv - sqrtf(dt) * eps;
        sh.u.d.h2[bb][d][n] = val;
      }
    }
    sh.u.d.z[bb][c][n] = f2b(val);
  }
  __syncthreads();

  // ---- diffusion hops: z[p+1][c][w] = sum_v z[p][c][v] * adj[w][v] ----
  // A past-row-end reads (k>=48) are multiplied by adjP zeros (v>=36) -> 0.
  for (int pstep = 0; pstep < 3; ++pstep) {
    for (int t = wave; t < BBB * 18; t += NWAVE) {
      int bb = t / 18, mt = (t % 18) / 3, nt = t % 3;
      v8f acc = {};
#pragma unroll
      for (int kk = 0; kk < 64; kk += 32) {
        v16bf a = ldA(&sh.u.d.z[bb][pstep * 96][0], SP, mt * 16, kk, lane);
        v16bf b = ldBT(&sh.u.d.adjP[0][0], 64, kk, nt * 16, lane);
        acc = wmma32(a, b, acc);
      }
      int n = nt * 16 + (lane & 15);
#pragma unroll
      for (int r = 0; r < 8; ++r)
        sh.u.d.z[bb][(pstep + 1) * 96 + mt * 16 + r + moff0][n] = f2b(acc[r]);
    }
    __syncthreads();
  }

  // ---- gates: u/c = W(64x384) x z(384x48); one z fragment feeds 8 WMMAs ----
  for (int unit = wave; unit < BBB * 3; unit += NWAVE) {
    int bb = unit / 3, nt = unit % 3;
    v8f au[4] = {{}, {}, {}, {}};
    v8f ac[4] = {{}, {}, {}, {}};
#pragma unroll 4
    for (int kk = 0; kk < CINZ; kk += 32) {
      v16bf bz = ldBs(&sh.u.d.z[bb][0][0], SP, kk, nt * 16, lane);
#pragma unroll
      for (int mt = 0; mt < 4; ++mt) {
        v16bf aw = ldA(&sh.u.d.uw[0][0], CINZ, mt * 16, kk, lane);
        au[mt] = wmma32(aw, bz, au[mt]);
        v16bf aw2 = ldA(&sh.u.d.cw[0][0], CINZ, mt * 16, kk, lane);
        ac[mt] = wmma32(aw2, bz, ac[mt]);
      }
    }
    int n = nt * 16 + (lane & 15);
    if (n < NNODE) {
#pragma unroll
      for (int mt = 0; mt < 4; ++mt) {
#pragma unroll
        for (int r = 0; r < 8; ++r) {
          int m = mt * 16 + r + moff0;
          float uu = 1.f / (1.f + __expf(-(au[mt][r] + p.u_b[m])));
          float cc = tanhf(ac[mt][r] + p.c_b[m]);
          float hh = sh.u.d.h2[bb][m][n];
          p.out[((size_t)(b0 + bb) * DMODEL + m) * NNODE + n] =
              uu * hh + (1.f - uu) * cc;
        }
      }
    }
  }
}

extern "C" void kernel_launch(void* const* d_in, const int* in_sizes, int n_in,
                              void* d_out, int out_size, void* d_ws, size_t ws_size,
                              hipStream_t stream) {
  (void)in_sizes; (void)n_in; (void)out_size; (void)d_ws; (void)ws_size;
  Params p;
  p.x   = (const float*)d_in[0];
  p.dt  = (const float*)d_in[1];
  p.h   = (const float*)d_in[2];
  p.adj = (const float*)d_in[3];
  for (int L = 0; L < 2; ++L) {
    int base = 4 + L * 12;
    p.in_w[L]  = (const float*)d_in[base + 0];
    p.in_b[L]  = (const float*)d_in[base + 1];
    p.out_w[L] = (const float*)d_in[base + 2];
    p.out_b[L] = (const float*)d_in[base + 3];
    p.ff1_w[L] = (const float*)d_in[base + 4];
    p.ff1_b[L] = (const float*)d_in[base + 5];
    p.ff2_w[L] = (const float*)d_in[base + 6];
    p.ff2_b[L] = (const float*)d_in[base + 7];
    p.ln1_g[L] = (const float*)d_in[base + 8];
    p.ln1_b[L] = (const float*)d_in[base + 9];
    p.ln2_g[L] = (const float*)d_in[base + 10];
    p.ln2_b[L] = (const float*)d_in[base + 11];
  }
  p.u_w = (const float*)d_in[28];
  p.u_b = (const float*)d_in[29];
  p.c_w = (const float*)d_in[30];
  p.c_b = (const float*)d_in[31];
  p.out = (float*)d_out;

  gcgru_fused_kernel<<<BSZ / BBB, NTHR, 0, stream>>>(p);
}
